// ContinuousHSpaceTeacher_4483945857523
// MI455X (gfx1250) — matrix operand order repaired and checked by reading, not compiled, not run on hardware
//
#include <hip/hip_runtime.h>

// ---------------------------------------------------------------------------
// Model constants (match reference)
// ---------------------------------------------------------------------------
#define Bc   32
#define Lc   512
#define Dc   1024
#define Hc   4
#define NLc  4
#define DKc  256
#define DVc  256
#define FFc  4096
#define GATE_NORM 16.0f

typedef __attribute__((ext_vector_type(16))) __bf16 bf16x16;
typedef __attribute__((ext_vector_type(8)))  __bf16 bf16x8;
typedef __attribute__((ext_vector_type(8)))  float  f32x8;
typedef __attribute__((ext_vector_type(4)))  unsigned int u32x4;
typedef __attribute__((ext_vector_type(4)))  int i32x4;
typedef __attribute__((ext_vector_type(8)))  int i32x8;

// ---------------------------------------------------------------------------
// TDM: 2-D tile load (bf16 elements) from global tensor into LDS.
// Descriptor bit layout per CDNA5 ISA 8.3/8.4 (D# group0 + group1).
//   tensor: [dim1 rows][dim0 cols], row stride = stride0 (elements)
//   tile:   [tile1 rows][tile0 cols], stored compacted at lds_addr
// 6-arg builtin form: (u32x4 g0, i32x8 g1, i32x4 g2, i32x4 g3, i32x8, i32 cpol)
// groups 2/3 and the extra operand are zero (2-D tensor, no gather/iterate).
// ---------------------------------------------------------------------------
__device__ __forceinline__ void tdm_load_2d_bf16(unsigned lds_addr,
                                                 const void* gptr,
                                                 unsigned dim0, unsigned dim1,
                                                 unsigned long long stride0,
                                                 unsigned tile0, unsigned tile1)
{
    unsigned long long ga = (unsigned long long)gptr;
    u32x4 g0;
    g0[0] = 1u;                                      // count=1, user descriptor
    g0[1] = lds_addr;                                // LDS byte address
    g0[2] = (unsigned)(ga & 0xffffffffull);          // global_addr[31:0]
    g0[3] = (unsigned)((ga >> 32) & 0x01ffffffull)   // global_addr[56:32]
          | 0x80000000u;                             // type=2 ("image")
    i32x8 g1;
    g1[0] = (int)(1u << 16);                         // data_size=1 (2 bytes), no mcast
    g1[1] = (int)((dim0 & 0xffffu) << 16);           // tensor_dim0[15:0]
    g1[2] = (int)((dim0 >> 16) | ((dim1 & 0xffffu) << 16));
    g1[3] = (int)((dim1 >> 16) | ((tile0 & 0xffffu) << 16));
    g1[4] = (int)(tile1 & 0xffffu);                  // tile_dim1 (tile_dim2 = 0)
    g1[5] = (int)(stride0 & 0xffffffffull);          // tensor_dim0_stride[31:0]
    g1[6] = (int)((unsigned)(stride0 >> 32) & 0xffffu);
    g1[7] = 0;
    i32x4 z4 = {0, 0, 0, 0};
    i32x8 z8 = {0, 0, 0, 0, 0, 0, 0, 0};
    __builtin_amdgcn_tensor_load_to_lds(g0, g1, z4, z4, z8, 0);
}

// ---------------------------------------------------------------------------
// WMMA GEMM:  C[32,N] = A[32,K](bf16) * W[K,N]  with W stored transposed
// as Wt[N][K] (bf16).  128 threads = 4 waves; workgroup owns a 128-column
// slab; each wave owns 32 columns (two 16-col WMMA subtiles) x two 16-row
// M tiles -> 4 accumulators, 4 WMMAs per staged k-step.
// MODE 0: C = acc * scale     MODE 1: C = resid + acc   (residual add)
// Double-buffered TDM staging of A-tile (32x32) and Wt-tile (128x32) in LDS.
// ---------------------------------------------------------------------------
template <int MODE>
__global__ __launch_bounds__(128) void gemm_wmma_bf16(
    const __bf16* __restrict__ A,    // [32, K]
    const __bf16* __restrict__ Wt,   // [N, K]
    float* __restrict__ C,           // [32, N]
    const float* __restrict__ resid, // [32, N] (MODE 1)
    int K, int N, float scale)
{
    __shared__ __bf16 lsA[2][32 * 32];    // [m][k] compact  (2 KB per buf)
    __shared__ __bf16 lsB[2][128 * 32];   // [n][k] compact  (8 KB per buf)

    const int tid  = threadIdx.x;
    const int wave = tid >> 5;
    const int lane = tid & 31;
    const int grp  = lane >> 4;          // 0: lanes 0-15, 1: lanes 16-31
    const int l16  = lane & 15;
    const int n0   = blockIdx.x * 128;
    const int KT   = K >> 5;             // K / 32

    f32x8 acc00 = {0.f, 0.f, 0.f, 0.f, 0.f, 0.f, 0.f, 0.f};
    f32x8 acc01 = acc00, acc10 = acc00, acc11 = acc00;

    auto issue = [&](int buf, int kt) {
        unsigned la = (unsigned)(unsigned long long)(&lsA[buf][0]);
        unsigned lb = (unsigned)(unsigned long long)(&lsB[buf][0]);
        // A tile: rows 0..31, cols kt*32..kt*32+31
        tdm_load_2d_bf16(la, A + (size_t)kt * 32,
                         (unsigned)K, 32u, (unsigned long long)K, 32u, 32u);
        // Wt tile: rows n0..n0+127, cols kt*32..kt*32+31
        tdm_load_2d_bf16(lb, Wt + (size_t)n0 * K + (size_t)kt * 32,
                         (unsigned)K, (unsigned)N, (unsigned long long)K, 32u, 128u);
    };

    if (wave == 0) issue(0, 0);

    for (int kt = 0; kt < KT; ++kt) {
        const int buf = kt & 1;
        if (wave == 0) __builtin_amdgcn_s_wait_tensorcnt(0);
        __syncthreads();                       // tile ready + prev buffer free
        if (wave == 0 && (kt + 1) < KT) issue(buf ^ 1, kt + 1);

        const __bf16* pa = &lsA[buf][0];
        const __bf16* pb = &lsB[buf][0];

        // A fragments, ISA 16-bit A layout:
        //   lanes 0-15 : K {0..7, 16..23};  lanes 16-31 : K {8..15, 24..31}
        bf16x8 a0lo = *(const bf16x8*)(pa + l16 * 32 + grp * 8);
        bf16x8 a0hi = *(const bf16x8*)(pa + l16 * 32 + 16 + grp * 8);
        bf16x8 a1lo = *(const bf16x8*)(pa + (l16 + 16) * 32 + grp * 8);
        bf16x8 a1hi = *(const bf16x8*)(pa + (l16 + 16) * 32 + 16 + grp * 8);
        bf16x16 af0 = __builtin_shufflevector(a0lo, a0hi,
            0,1,2,3,4,5,6,7,8,9,10,11,12,13,14,15);
        bf16x16 af1 = __builtin_shufflevector(a1lo, a1hi,
            0,1,2,3,4,5,6,7,8,9,10,11,12,13,14,15);

        // B fragments: column n = n0 + wave*32 + ns*16 + l16;
        // lanes 0-15 K {0..7, 16..23}, lanes 16-31 K {8..15, 24..31} -> 16B LDS loads.
        const int nr0 = wave * 32 + l16;
        bf16x8 b0lo = *(const bf16x8*)(pb + nr0 * 32 + grp * 8);
        bf16x8 b0hi = *(const bf16x8*)(pb + nr0 * 32 + 16 + grp * 8);
        bf16x16 bf0 = __builtin_shufflevector(b0lo, b0hi,
            0,1,2,3,4,5,6,7,8,9,10,11,12,13,14,15);
        const int nr1 = nr0 + 16;
        bf16x8 b1lo = *(const bf16x8*)(pb + nr1 * 32 + grp * 8);
        bf16x8 b1hi = *(const bf16x8*)(pb + nr1 * 32 + 16 + grp * 8);
        bf16x16 bf1 = __builtin_shufflevector(b1lo, b1hi,
            0,1,2,3,4,5,6,7,8,9,10,11,12,13,14,15);

        acc00 = __builtin_amdgcn_wmma_f32_16x16x32_bf16(
                    false, af0, false, bf0, (short)0, acc00, false, false);
        acc10 = __builtin_amdgcn_wmma_f32_16x16x32_bf16(
                    false, af1, false, bf0, (short)0, acc10, false, false);
        acc01 = __builtin_amdgcn_wmma_f32_16x16x32_bf16(
                    false, af0, false, bf1, (short)0, acc01, false, false);
        acc11 = __builtin_amdgcn_wmma_f32_16x16x32_bf16(
                    false, af1, false, bf1, (short)0, acc11, false, false);
    }

    // C/D layout: VGPR r -> M = r + 8*grp (acc0*: M 0..15, acc1*: M 16..31),
    // N = lane%16 within each 16-column subtile.
    const int na = n0 + wave * 32 + l16;
    const int nb = na + 16;
    #pragma unroll
    for (int r = 0; r < 8; ++r) {
        const int m = grp * 8 + r;
        if (MODE == 0) {
            C[(size_t)m * N + na]        = acc00[r] * scale;
            C[(size_t)(m + 16) * N + na] = acc10[r] * scale;
            C[(size_t)m * N + nb]        = acc01[r] * scale;
            C[(size_t)(m + 16) * N + nb] = acc11[r] * scale;
        } else {
            C[(size_t)m * N + na]        = resid[(size_t)m * N + na] + acc00[r];
            C[(size_t)(m + 16) * N + na] = resid[(size_t)(m + 16) * N + na] + acc10[r];
            C[(size_t)m * N + nb]        = resid[(size_t)m * N + nb] + acc01[r];
            C[(size_t)(m + 16) * N + nb] = resid[(size_t)(m + 16) * N + nb] + acc11[r];
        }
    }
}

// ---------------------------------------------------------------------------
// Helpers
// ---------------------------------------------------------------------------
__device__ __forceinline__ float block_sum(float v, float* red)
{
    __syncthreads();
    for (int off = 16; off > 0; off >>= 1) v += __shfl_down(v, off);
    const int wid = threadIdx.x >> 5;
    if ((threadIdx.x & 31) == 0) red[wid] = v;
    __syncthreads();
    if (threadIdx.x == 0) {
        float t = 0.f;
        const int nw = (blockDim.x + 31) >> 5;
        for (int i = 0; i < nw; ++i) t += red[i];
        red[0] = t;
    }
    __syncthreads();
    return red[0];
}

// cast + transpose: W[K][N] f32 -> Wt[N][K] bf16
__global__ __launch_bounds__(256) void cast_transpose(
    const float* __restrict__ W, __bf16* __restrict__ Wt, int Kd, int Nd)
{
    size_t i = (size_t)blockIdx.x * 256 + threadIdx.x;
    size_t tot = (size_t)Kd * Nd;
    if (i < tot) {
        int nrow = (int)(i / Kd);
        int kcol = (int)(i % Kd);
        Wt[i] = (__bf16)W[(size_t)kcol * Nd + nrow];
    }
}

// x = rmsnorm(hidden, w) -> bf16
__global__ __launch_bounds__(256) void rmsnorm_to_bf16(
    const float* __restrict__ h, const float* __restrict__ w,
    __bf16* __restrict__ xb)
{
    __shared__ float red[8];
    const int b = blockIdx.x;
    const float* row = h + (size_t)b * Dc;
    float ss = 0.f;
    for (int d = threadIdx.x; d < Dc; d += 256) { float v = row[d]; ss += v * v; }
    float tot = block_sum(ss, red);
    float inv = rsqrtf(tot / (float)Dc + 1e-6f);
    for (int d = threadIdx.x; d < Dc; d += 256)
        xb[(size_t)b * Dc + d] = (__bf16)(row[d] * inv * w[d]);
}

// out[:,0] = hidden = l2norm(h0) * sqrt(D)
__global__ __launch_bounds__(256) void init_h0(
    const float* __restrict__ h0, float* __restrict__ out,
    float* __restrict__ hidden)
{
    __shared__ float red[8];
    const int b = blockIdx.x;
    const float* row = h0 + (size_t)b * Dc;
    float ss = 0.f;
    for (int d = threadIdx.x; d < Dc; d += 256) { float v = row[d]; ss += v * v; }
    float tot = block_sum(ss, red);
    float s = rsqrtf(tot + 1e-12f) * 32.0f;   // sqrt(1024) = 32
    for (int d = threadIdx.x; d < Dc; d += 256) {
        float v = row[d] * s;
        out[((size_t)b * Lc + 0) * Dc + d] = v;
        hidden[(size_t)b * Dc + d] = v;
    }
}

// gk = log_sigmoid((x @ Wa) @ Wb) / 16
__global__ __launch_bounds__(256) void gk_lowrank(
    const __bf16* __restrict__ xb, const float* __restrict__ Wa,
    const float* __restrict__ Wb, float* __restrict__ gk)
{
    __shared__ float part[16][17];
    __shared__ float t16[16];
    const int b = blockIdx.x;
    const int r = threadIdx.x & 15;
    const int c = threadIdx.x >> 4;
    float s = 0.f;
    for (int d = c * 64; d < c * 64 + 64; ++d)
        s += (float)xb[(size_t)b * Dc + d] * Wa[(size_t)d * 16 + r];
    part[c][r] = s;
    __syncthreads();
    if (threadIdx.x < 16) {
        float t = 0.f;
        for (int i = 0; i < 16; ++i) t += part[i][threadIdx.x];
        t16[threadIdx.x] = t;
    }
    __syncthreads();
    for (int n = threadIdx.x; n < Dc; n += 256) {
        float z = 0.f;
        #pragma unroll
        for (int rr = 0; rr < 16; ++rr) z += t16[rr] * Wb[(size_t)rr * Dc + n];
        float ls = fminf(z, 0.f) - log1pf(expf(-fabsf(z)));  // stable logsigmoid
        gk[(size_t)b * Dc + n] = ls / GATE_NORM;
    }
}

// GLA state update + output + RMSNormSwishGate, one workgroup per (b,h)
__global__ __launch_bounds__(256) void gla_state(
    float* __restrict__ S,            // [B*H, DK, DV] (this layer)
    const float* __restrict__ q,      // [B, D] (already * DK^-0.5)
    const float* __restrict__ k,
    const float* __restrict__ v,
    const float* __restrict__ gk,
    const float* __restrict__ gproj,  // [B, D]
    const float* __restrict__ gnw,    // [DV]
    __bf16* __restrict__ ofinal)      // [B, D]
{
    __shared__ float qs[DKc], ks[DKc], gs[DKc];
    __shared__ float red[8];
    const int bh = blockIdx.x;
    const int b = bh >> 2, h = bh & 3;
    const int tid = threadIdx.x;
    const size_t base = (size_t)b * Dc + h * DKc;
    qs[tid] = q[base + tid];
    ks[tid] = k[base + tid];
    gs[tid] = expf(gk[base + tid]);
    __syncthreads();
    const float vv = v[base + tid];
    float* Sp = S + (size_t)bh * DKc * DVc;
    float acc = 0.f;
    for (int kk = 0; kk < DKc; ++kk) {
        float s = Sp[(size_t)kk * DVc + tid];
        s = gs[kk] * s + ks[kk] * vv;
        Sp[(size_t)kk * DVc + tid] = s;
        acc += qs[kk] * s;
    }
    float tot = block_sum(acc * acc, red);
    float inv = rsqrtf(tot / (float)DVc + 1e-6f);
    float g = gproj[base + tid];
    float sw = g * (1.f / (1.f + expf(-g)));
    ofinal[base + tid] = (__bf16)(acc * inv * gnw[tid] * sw);
}

// act = silu(gate) * up -> bf16
__global__ __launch_bounds__(256) void silu_mul(
    const float* __restrict__ gt, const float* __restrict__ ut,
    __bf16* __restrict__ outb, int n)
{
    int i = blockIdx.x * 256 + threadIdx.x;
    if (i < n) {
        float g = gt[i];
        float s = g * (1.f / (1.f + expf(-g)));
        outb[i] = (__bf16)(s * ut[i]);
    }
}

// h_t = hidden + noise*sigma; out[:,t] = h_t; hidden = l2norm(h_t)*sqrt(D)
__global__ __launch_bounds__(256) void finalize_step(
    float* __restrict__ hidden, const float* __restrict__ noise_t,
    const float* __restrict__ sigma, float* __restrict__ out, int t)
{
    __shared__ float hrow[Dc];
    __shared__ float red[8];
    const int b = blockIdx.x;
    const float sg = sigma[0];
    float ss = 0.f;
    for (int d = threadIdx.x; d < Dc; d += 256) {
        float v = hidden[(size_t)b * Dc + d] + noise_t[(size_t)b * Dc + d] * sg;
        hrow[d] = v;
        out[((size_t)b * Lc + t) * Dc + d] = v;
        ss += v * v;
    }
    float tot = block_sum(ss, red);
    float s = rsqrtf(tot + 1e-12f) * 32.0f;
    for (int d = threadIdx.x; d < Dc; d += 256)
        hidden[(size_t)b * Dc + d] = hrow[d] * s;
}

// ---------------------------------------------------------------------------
// Host orchestration
// ---------------------------------------------------------------------------
extern "C" void kernel_launch(void* const* d_in, const int* in_sizes, int n_in,
                              void* d_out, int out_size, void* d_ws, size_t ws_size,
                              hipStream_t stream)
{
    (void)in_sizes; (void)n_in; (void)out_size; (void)ws_size;
    const float* h0    = (const float*)d_in[0];
    const float* noise = (const float*)d_in[1];
    const float* sigma = (const float*)d_in[2];
    const float* anw   = (const float*)d_in[3];
    const float* Wq    = (const float*)d_in[4];
    const float* Wk    = (const float*)d_in[5];
    const float* Wv    = (const float*)d_in[6];
    const float* Wgka  = (const float*)d_in[7];
    const float* Wgkb  = (const float*)d_in[8];
    const float* Wg    = (const float*)d_in[9];
    const float* gnw   = (const float*)d_in[10];
    const float* Wo    = (const float*)d_in[11];
    const float* mnw   = (const float*)d_in[12];
    const float* Wgate = (const float*)d_in[13];
    const float* Wup   = (const float*)d_in[14];
    const float* Wdown = (const float*)d_in[15];
    float* out = (float*)d_out;

    char* w = (char*)d_ws;
    size_t off = 0;
    auto alloc = [&](size_t bytes) {
        void* p = w + off;
        off = (off + bytes + 255) & ~(size_t)255;
        return p;
    };
    const size_t DD = (size_t)Dc * Dc, DFF = (size_t)Dc * FFc;
    float*  S       = (float*) alloc((size_t)NLc * Bc * Hc * DKc * DVc * 4);
    __bf16* WqT     = (__bf16*)alloc(NLc * DD * 2);
    __bf16* WkT     = (__bf16*)alloc(NLc * DD * 2);
    __bf16* WvT     = (__bf16*)alloc(NLc * DD * 2);
    __bf16* WgT     = (__bf16*)alloc(NLc * DD * 2);
    __bf16* WoT     = (__bf16*)alloc(NLc * DD * 2);
    __bf16* WgateT  = (__bf16*)alloc(NLc * DFF * 2);
    __bf16* WupT    = (__bf16*)alloc(NLc * DFF * 2);
    __bf16* WdownT  = (__bf16*)alloc(NLc * DFF * 2);
    float*  hidden  = (float*) alloc((size_t)Bc * Dc * 4);
    __bf16* xb      = (__bf16*)alloc((size_t)Bc * Dc * 2);
    __bf16* yb      = (__bf16*)alloc((size_t)Bc * Dc * 2);
    __bf16* ob      = (__bf16*)alloc((size_t)Bc * Dc * 2);
    float*  qbuf    = (float*) alloc((size_t)Bc * Dc * 4);
    float*  kbuf    = (float*) alloc((size_t)Bc * Dc * 4);
    float*  vbuf    = (float*) alloc((size_t)Bc * Dc * 4);
    float*  gbuf    = (float*) alloc((size_t)Bc * Dc * 4);
    float*  gkbuf   = (float*) alloc((size_t)Bc * Dc * 4);
    float*  ggbuf   = (float*) alloc((size_t)Bc * FFc * 4);
    float*  gubuf   = (float*) alloc((size_t)Bc * FFc * 4);
    __bf16* actb    = (__bf16*)alloc((size_t)Bc * FFc * 2);

    // zero the GLA state (deterministic across calls)
    (void)hipMemsetAsync(S, 0, (size_t)NLc * Bc * Hc * DKc * DVc * 4, stream);

    // weight prep: f32 [K][N] -> bf16 [N][K]
    const int gDD  = (int)((DD  + 255) / 256);
    const int gDFF = (int)((DFF + 255) / 256);
    for (int l = 0; l < NLc; ++l) {
        cast_transpose<<<gDD, 256, 0, stream>>>(Wq    + l * DD,  WqT    + l * DD,  Dc, Dc);
        cast_transpose<<<gDD, 256, 0, stream>>>(Wk    + l * DD,  WkT    + l * DD,  Dc, Dc);
        cast_transpose<<<gDD, 256, 0, stream>>>(Wv    + l * DD,  WvT    + l * DD,  Dc, Dc);
        cast_transpose<<<gDD, 256, 0, stream>>>(Wg    + l * DD,  WgT    + l * DD,  Dc, Dc);
        cast_transpose<<<gDD, 256, 0, stream>>>(Wo    + l * DD,  WoT    + l * DD,  Dc, Dc);
        cast_transpose<<<gDFF, 256, 0, stream>>>(Wgate + l * DFF, WgateT + l * DFF, Dc, FFc);
        cast_transpose<<<gDFF, 256, 0, stream>>>(Wup   + l * DFF, WupT   + l * DFF, Dc, FFc);
        cast_transpose<<<gDFF, 256, 0, stream>>>(Wdown + l * DFF, WdownT + l * DFF, FFc, Dc);
    }

    init_h0<<<Bc, 256, 0, stream>>>(h0, out, hidden);

    const float qscale = 0.0625f;  // DK^-0.5 = 1/16
    for (int t = 1; t < Lc; ++t) {
        for (int l = 0; l < NLc; ++l) {
            const size_t sl = (size_t)l * Bc * Hc * DKc * DVc;
            rmsnorm_to_bf16<<<Bc, 256, 0, stream>>>(hidden, anw + (size_t)l * Dc, xb);
            gemm_wmma_bf16<0><<<Dc / 128, 128, 0, stream>>>(xb, WqT + l * DD, qbuf, nullptr, Dc, Dc, qscale);
            gemm_wmma_bf16<0><<<Dc / 128, 128, 0, stream>>>(xb, WkT + l * DD, kbuf, nullptr, Dc, Dc, 1.f);
            gemm_wmma_bf16<0><<<Dc / 128, 128, 0, stream>>>(xb, WvT + l * DD, vbuf, nullptr, Dc, Dc, 1.f);
            gemm_wmma_bf16<0><<<Dc / 128, 128, 0, stream>>>(xb, WgT + l * DD, gbuf, nullptr, Dc, Dc, 1.f);
            gk_lowrank<<<Bc, 256, 0, stream>>>(xb, Wgka + (size_t)l * Dc * 16,
                                               Wgkb + (size_t)l * 16 * Dc, gkbuf);
            gla_state<<<Bc * Hc, 256, 0, stream>>>(S + sl, qbuf, kbuf, vbuf, gkbuf,
                                                   gbuf, gnw + (size_t)l * DVc, ob);
            gemm_wmma_bf16<1><<<Dc / 128, 128, 0, stream>>>(ob, WoT + l * DD, hidden, hidden, Dc, Dc, 1.f);
            rmsnorm_to_bf16<<<Bc, 256, 0, stream>>>(hidden, mnw + (size_t)l * Dc, yb);
            gemm_wmma_bf16<0><<<FFc / 128, 128, 0, stream>>>(yb, WgateT + l * DFF, ggbuf, nullptr, Dc, FFc, 1.f);
            gemm_wmma_bf16<0><<<FFc / 128, 128, 0, stream>>>(yb, WupT   + l * DFF, gubuf, nullptr, Dc, FFc, 1.f);
            silu_mul<<<(Bc * FFc + 255) / 256, 256, 0, stream>>>(ggbuf, gubuf, actb, Bc * FFc);
            gemm_wmma_bf16<1><<<Dc / 128, 128, 0, stream>>>(actb, WdownT + l * DFF, hidden, hidden, FFc, Dc, 1.f);
        }
        finalize_step<<<Bc, 256, 0, stream>>>(hidden, noise + (size_t)(t - 1) * Bc * Dc,
                                              sigma, out, t);
    }
}
